// STSA_13881334300697
// MI455X (gfx1250) — matrix operand order repaired and checked by reading, not compiled, not run on hardware
//
#include <hip/hip_runtime.h>

// ---------------------------------------------------------------------------
// STSA for MI455X (gfx1250): all heavy math through v_wmma_f32_16x16x32_f16,
// LDS staging through GLOBAL_LOAD_ASYNC_TO_LDS_B128 where available.
// Pipeline:
//   prep:    x -> f16, pack W6 (768x128), Wc (128x256), biases
//   gemm:    QKV = xh @ W6^T + b6           (f16 out, 65536 x 768)
//   flash:   temporal attn (seq 64) + spatial attn (seq 256, gathered KV)
//            -> attn buffer (65536 x 256 f16; cols 0..127 temporal, 128..255 spatial)
//   gemm:    y = attn @ Wc^T + bo           (f32 out, straight into d_out)
// ---------------------------------------------------------------------------

typedef __attribute__((ext_vector_type(16))) _Float16 v16h;
typedef __attribute__((ext_vector_type(8)))  float    v8f;
typedef int v4i_vs __attribute__((vector_size(16)));   // matches builtin param type

#ifndef __has_builtin
#define __has_builtin(x) 0
#endif
#if __has_builtin(__builtin_amdgcn_global_load_async_to_lds_b128)
#define HAVE_ASYNC_LDS 1
#else
#define HAVE_ASYNC_LDS 0
#endif

// 16-byte global -> LDS copy. Async DMA path on gfx1250 (ASYNCcnt-tracked),
// falls back to a VGPR round-trip if the builtin is unavailable.
static __device__ inline void cp16_async(const _Float16* g, _Float16* l) {
#if HAVE_ASYNC_LDS
    __builtin_amdgcn_global_load_async_to_lds_b128(
        (__attribute__((address_space(1))) v4i_vs*)g,
        (__attribute__((address_space(3))) v4i_vs*)l, 0, 0);
#else
    *(uint4*)l = *(const uint4*)g;
#endif
}
static __device__ inline void async_join() {
#if HAVE_ASYNC_LDS
#if __has_builtin(__builtin_amdgcn_s_wait_asynccnt)
    __builtin_amdgcn_s_wait_asynccnt(0);
#else
    asm volatile("s_wait_asynccnt 0" ::: "memory");
#endif
#endif
}

union FragU { v16h v; uint4 u[2]; };

static __device__ inline v8f wmma_f16(v16h a, v16h b, v8f c) {
    return __builtin_amdgcn_wmma_f32_16x16x32_f16(
        /*neg_a=*/false, a, /*neg_b=*/false, b,
        /*c_mod=*/(short)0, c, /*reuse_a=*/false, /*reuse_b=*/false);
}

// A-fragment: 16x32 f16. lane&15 = row M; lane>>4 selects K {0..15} vs {16..31}.
static __device__ inline v16h ld_a_frag(const _Float16* row, int khalf) {
    FragU f;
    f.u[0] = *(const uint4*)(row + khalf * 16);
    f.u[1] = *(const uint4*)(row + khalf * 16 + 8);
    return f.v;
}
// A-fragment with K=16 real, 16..31 zero (hd=16 padded to 32)
static __device__ inline v16h ld_a_frag_k16(const _Float16* row, int khalf) {
    FragU f;
    if (khalf == 0) {
        f.u[0] = *(const uint4*)(row);
        f.u[1] = *(const uint4*)(row + 8);
    } else {
        f.u[0] = make_uint4(0u, 0u, 0u, 0u);
        f.u[1] = make_uint4(0u, 0u, 0u, 0u);
    }
    return f.v;
}
// B-fragment: 32x16 f16 stored as Bs[n][k] rows. lane&15 = col N;
// lane>>4 selects K {0..15} vs {16..31}: 32 contiguous bytes.
static __device__ inline v16h ld_b_frag(const _Float16* row, int khalf) {
    FragU f;
    const uint4* p = (const uint4*)(row + khalf * 16);
    f.u[0] = p[0];
    f.u[1] = p[1];
    return f.v;
}
// B-fragment with K=16 real (khalf==1 half is the zero padding)
static __device__ inline v16h ld_b_frag_k16(const _Float16* row, int khalf) {
    FragU f;
    if (khalf == 0) {
        const uint4* p = (const uint4*)row;
        f.u[0] = p[0];
        f.u[1] = p[1];
    } else {
        f.u[0] = make_uint4(0u, 0u, 0u, 0u);
        f.u[1] = make_uint4(0u, 0u, 0u, 0u);
    }
    return f.v;
}

// ---------------------------------------------------------------------------
// Prep kernels
// ---------------------------------------------------------------------------
__global__ void cvt_x_kernel(const float* __restrict__ x, _Float16* __restrict__ xh, int n) {
    int i = blockIdx.x * blockDim.x + threadIdx.x;
    if (i < n) xh[i] = (_Float16)x[i];
}

__global__ void prep_w_kernel(const float* __restrict__ t_w_in, const float* __restrict__ s_w_in,
                              const float* __restrict__ t_b_in, const float* __restrict__ s_b_in,
                              const float* __restrict__ t_w_out, const float* __restrict__ s_w_out,
                              const float* __restrict__ t_b_out, const float* __restrict__ s_b_out,
                              _Float16* __restrict__ W6h, _Float16* __restrict__ Wch,
                              float* __restrict__ b6, float* __restrict__ bo) {
    int i = blockIdx.x * blockDim.x + threadIdx.x;
    if (i < 768 * 128) {           // W6h[row][k], rows: 0..383 = t_w_in, 384..767 = s_w_in
        W6h[i] = (_Float16)((i < 384 * 128) ? t_w_in[i] : s_w_in[i - 384 * 128]);
    }
    if (i < 128 * 256) {           // Wch[o][k], k<128 -> t_w_out, k>=128 -> s_w_out
        int o = i >> 8, k = i & 255;
        Wch[i] = (_Float16)((k < 128) ? t_w_out[o * 128 + k] : s_w_out[o * 128 + (k - 128)]);
    }
    if (i < 768) b6[i] = (i < 384) ? t_b_in[i] : s_b_in[i - 384];
    if (i < 128) bo[i] = t_b_out[i] + s_b_out[i];
}

// ---------------------------------------------------------------------------
// Tiled WMMA GEMM: C(MxN) = A(MxK) @ B(NxK)^T + bias[N]
// 128x128 block tile, BK=64. 8 waves arranged 4(M) x 2(N); each wave owns a
// 32x64 subtile = 2x4 wmma tiles. LDS staging via async-to-LDS DMA.
// ---------------------------------------------------------------------------
template <int OUT_F32>
__global__ __launch_bounds__(256) void gemm_wmma_kernel(
        const _Float16* __restrict__ A, const _Float16* __restrict__ B,
        const float* __restrict__ bias, void* __restrict__ Cp,
        int M, int N, int K) {
    __shared__ _Float16 As[128 * 72];
    __shared__ _Float16 Bs[128 * 72];

    const int tid   = threadIdx.x;
    const int wave  = tid >> 5;
    const int lane  = tid & 31;
    const int n16   = lane & 15;
    const int khalf = lane >> 4;
    const int wm    = wave & 3;   // 0..3 -> M offset wm*32
    const int wn    = wave >> 2;  // 0..1 -> N offset wn*64
    const int m0    = blockIdx.x * 128;
    const int n0    = blockIdx.y * 128;

    v8f acc[2][4] = {};

    for (int k0 = 0; k0 < K; k0 += 64) {
        // stage A, B tiles (128 rows x 64 halves each): 1024 b128 copies apiece
        #pragma unroll
        for (int it = 0; it < 4; ++it) {
            int task = tid + it * 256;
            int row  = task >> 3;
            int c8   = (task & 7) * 8;
            cp16_async(&A[(size_t)(m0 + row) * K + k0 + c8], &As[row * 72 + c8]);
            cp16_async(&B[(size_t)(n0 + row) * K + k0 + c8], &Bs[row * 72 + c8]);
        }
        if (k0 + 64 < K)
            __builtin_prefetch(&A[(size_t)(m0 + (tid >> 1)) * K + k0 + 64], 0, 1);
        async_join();
        __syncthreads();

        #pragma unroll
        for (int ks = 0; ks < 64; ks += 32) {
            v16h af[2], bf[4];
            #pragma unroll
            for (int mt = 0; mt < 2; ++mt)
                af[mt] = ld_a_frag(&As[(wm * 32 + mt * 16 + n16) * 72 + ks], khalf);
            #pragma unroll
            for (int nt = 0; nt < 4; ++nt)
                bf[nt] = ld_b_frag(&Bs[(wn * 64 + nt * 16 + n16) * 72 + ks], khalf);
            #pragma unroll
            for (int mt = 0; mt < 2; ++mt)
                #pragma unroll
                for (int nt = 0; nt < 4; ++nt)
                    acc[mt][nt] = wmma_f16(af[mt], bf[nt], acc[mt][nt]);
        }
        __syncthreads();
    }

    // epilogue: D tile row = r + 8*khalf, col = n16
    #pragma unroll
    for (int mt = 0; mt < 2; ++mt) {
        #pragma unroll
        for (int nt = 0; nt < 4; ++nt) {
            #pragma unroll
            for (int r = 0; r < 8; ++r) {
                int gm = m0 + wm * 32 + mt * 16 + r + 8 * khalf;
                int gn = n0 + wn * 64 + nt * 16 + n16;
                float v = acc[mt][nt][r] + bias[gn];
                if (OUT_F32) ((float*)Cp)[(size_t)gm * N + gn] = v;
                else         ((_Float16*)Cp)[(size_t)gm * N + gn] = (_Float16)v;
            }
        }
    }
}

// ---------------------------------------------------------------------------
// Flash attention kernel. Block = (b, m, query-tile of 16); wave = head (8 waves).
// mode 0: temporal (Kl=64), mode 1: spatial (Kl=256, gathered neighbor tokens).
// Scores: WMMA with hd=16 zero-padded to K=32. Online softmax across 64-key
// chunks; cross-lane row reductions batched 8-wide per xor step so the
// ds_bpermute waves pipeline instead of serializing on s_wait_dscnt.
// ---------------------------------------------------------------------------
__global__ __launch_bounds__(256) void flash_kernel(
        const _Float16* __restrict__ qkv,   // [65536][768]
        const int* __restrict__ route,      // [256][4]
        _Float16* __restrict__ attn,        // [65536][256]
        int mode) {
    __shared__ _Float16 Qs[16 * 136];       // [qrow][128]
    __shared__ _Float16 Ks[64 * 136];       // [key ][128]
    __shared__ _Float16 Vst[128 * 72];      // [col ][key]  (V transposed)
    __shared__ _Float16 Pws[8 * 16 * 72];   // per-wave P: [qrow][key]

    const int tid   = threadIdx.x;
    const int wave  = tid >> 5;             // head
    const int lane  = tid & 31;
    const int n16   = lane & 15;
    const int khalf = lane >> 4;
    const int b     = blockIdx.x >> 8;
    const int m     = blockIdx.x & 255;
    const int qt    = blockIdx.y;
    const int h     = wave;

    const int Kl   = mode ? 256 : 64;
    const int qoff = mode ? 384 : 0;
    const int koff = mode ? 512 : 128;
    const int voff = mode ? 640 : 256;
    const int aoff = mode ? 128 : 0;
    const float scale = 0.25f;              // 1/sqrt(16)

    // stage Q tile (16 x 128): one b128 copy per thread (joined with chunk 0)
    {
        int i  = tid >> 4;
        int c8 = (tid & 15) * 8;
        int q  = qt * 16 + i;
        size_t tok = ((size_t)(b * 64 + q) * 256 + m);
        cp16_async(&qkv[tok * 768 + qoff + c8], &Qs[i * 136 + c8]);
    }

    float rmax[8], rsum[8];
    #pragma unroll
    for (int r = 0; r < 8; ++r) { rmax[r] = -1e30f; rsum[r] = 0.0f; }
    v8f oacc = {};

    _Float16* Pw = &Pws[h * 16 * 72];

    for (int j0 = 0; j0 < Kl; j0 += 64) {
        // ---- stage K chunk (64x128, async) and V^T chunk (128x64, transpose) ----
        __syncthreads();
        #pragma unroll
        for (int it = 0; it < 4; ++it) {
            int task = tid + it * 256;       // 1024 tasks = 64 keys x 16 col-segments
            int jj   = task >> 4;
            int c8   = (task & 15) * 8;
            int j    = j0 + jj;
            size_t tokk;
            if (mode == 0) {
                tokk = ((size_t)(b * 64 + j) * 256 + m);
            } else {
                int t2  = j >> 2;
                int kk  = j & 3;
                int nbr = route[m * 4 + kk];
                tokk = ((size_t)(b * 64 + t2) * 256 + nbr);
            }
            cp16_async(&qkv[tokk * 768 + koff + c8], &Ks[jj * 136 + c8]);
            uint4 vraw = *(const uint4*)&qkv[tokk * 768 + voff + c8];
            const _Float16* vp = (const _Float16*)&vraw;
            #pragma unroll
            for (int e = 0; e < 8; ++e)
                Vst[(c8 + e) * 72 + jj] = vp[e];
        }
        async_join();
        __syncthreads();

        // ---- scores: S (16 x 64) = Qh (16x16 pad32) @ Kh^T ----
        v16h qf = ld_a_frag_k16(&Qs[n16 * 136 + h * 16], khalf);
        v8f sc[4];
        #pragma unroll
        for (int nt = 0; nt < 4; ++nt) {
            v16h bf = ld_b_frag_k16(&Ks[(nt * 16 + n16) * 136 + h * 16], khalf);
            v8f z = {};
            sc[nt] = wmma_f16(qf, bf, z);
        }
        #pragma unroll
        for (int nt = 0; nt < 4; ++nt) sc[nt] *= scale;

        // ---- online softmax update (row = r + 8*khalf, shared by 16 lanes/half) ----
        float mc[8];
        #pragma unroll
        for (int r = 0; r < 8; ++r)
            mc[r] = fmaxf(fmaxf(sc[0][r], sc[1][r]), fmaxf(sc[2][r], sc[3][r]));
        #pragma unroll
        for (int mask = 1; mask <= 8; mask <<= 1) {
            float t[8];
            #pragma unroll
            for (int r = 0; r < 8; ++r) t[r] = __shfl_xor(mc[r], mask, 32);
            #pragma unroll
            for (int r = 0; r < 8; ++r) mc[r] = fmaxf(mc[r], t[r]);
        }
        float al[8], ps[8];
        #pragma unroll
        for (int r = 0; r < 8; ++r) {
            float nm = fmaxf(rmax[r], mc[r]);
            al[r] = __expf(rmax[r] - nm);
            rmax[r] = nm;
            float s = 0.0f;
            #pragma unroll
            for (int nt = 0; nt < 4; ++nt) {
                float p = __expf(sc[nt][r] - nm);
                sc[nt][r] = p;
                s += p;
            }
            ps[r] = s;
        }
        #pragma unroll
        for (int mask = 1; mask <= 8; mask <<= 1) {
            float t[8];
            #pragma unroll
            for (int r = 0; r < 8; ++r) t[r] = __shfl_xor(ps[r], mask, 32);
            #pragma unroll
            for (int r = 0; r < 8; ++r) ps[r] += t[r];
        }
        #pragma unroll
        for (int r = 0; r < 8; ++r) {
            rsum[r] = rsum[r] * al[r] + ps[r];
            oacc[r] *= al[r];
        }

        // ---- stage P (16x64) into per-wave LDS, then O += P @ V ----
        #pragma unroll
        for (int r = 0; r < 8; ++r)
            #pragma unroll
            for (int nt = 0; nt < 4; ++nt)
                Pw[(r + 8 * khalf) * 72 + nt * 16 + n16] = (_Float16)sc[nt][r];
        asm volatile("s_wait_dscnt 0" ::: "memory");

        #pragma unroll
        for (int ks = 0; ks < 2; ++ks) {
            v16h pf = ld_a_frag(&Pw[n16 * 72 + ks * 32], khalf);
            v16h vf = ld_b_frag(&Vst[(h * 16 + n16) * 72 + ks * 32], khalf);
            oacc = wmma_f16(pf, vf, oacc);
        }
    }

    // ---- normalize + store ----
    #pragma unroll
    for (int r = 0; r < 8; ++r) {
        float o = oacc[r] / rsum[r];
        int q = qt * 16 + r + 8 * khalf;
        size_t tok = ((size_t)(b * 64 + q) * 256 + m);
        attn[tok * 256 + aoff + h * 16 + n16] = (_Float16)o;
    }
}

// ---------------------------------------------------------------------------
// Launch
// ---------------------------------------------------------------------------
extern "C" void kernel_launch(void* const* d_in, const int* in_sizes, int n_in,
                              void* d_out, int out_size, void* d_ws, size_t ws_size,
                              hipStream_t stream) {
    (void)in_sizes; (void)n_in; (void)out_size; (void)ws_size;
    const float* x        = (const float*)d_in[0];
    const int*   route    = (const int*)  d_in[1];
    const float* t_w_in   = (const float*)d_in[2];
    const float* t_b_in   = (const float*)d_in[3];
    const float* t_w_out  = (const float*)d_in[4];
    const float* t_b_out  = (const float*)d_in[5];
    const float* s_w_in   = (const float*)d_in[6];
    const float* s_b_in   = (const float*)d_in[7];
    const float* s_w_out  = (const float*)d_in[8];
    const float* s_b_out  = (const float*)d_in[9];

    const int NTOK = 4 * 64 * 256;          // 65536 tokens
    char* ws = (char*)d_ws;
    _Float16* xh    = (_Float16*)(ws);                               // 16.8 MB
    _Float16* qkvh  = (_Float16*)(ws + 16777216);                    // 100.7 MB
    _Float16* attn  = (_Float16*)(ws + 16777216 + 100663296);        // 33.6 MB
    char* tail = ws + 16777216 + 100663296 + 33554432;
    _Float16* W6h   = (_Float16*)(tail);                             // 768x128
    _Float16* Wch   = (_Float16*)(tail + 196608);                    // 128x256
    float*    b6    = (float*)   (tail + 196608 + 65536);            // 768
    float*    bo    = (float*)   (tail + 196608 + 65536 + 4096);     // 128

    // prep
    cvt_x_kernel<<<(NTOK * 128) / 256, 256, 0, stream>>>(x, xh, NTOK * 128);
    prep_w_kernel<<<(768 * 128 + 255) / 256, 256, 0, stream>>>(
        t_w_in, s_w_in, t_b_in, s_b_in, t_w_out, s_w_out, t_b_out, s_b_out,
        W6h, Wch, b6, bo);

    // fused QKV projection: (65536 x 128) @ (768 x 128)^T -> f16
    gemm_wmma_kernel<0><<<dim3(NTOK / 128, 6), 256, 0, stream>>>(
        xh, W6h, b6, (void*)qkvh, NTOK, 768, 128);

    // temporal + spatial attention (block = (b,m) x query-tile, wave = head)
    flash_kernel<<<dim3(1024, 4), 256, 0, stream>>>(qkvh, route, attn, 0);
    flash_kernel<<<dim3(1024, 4), 256, 0, stream>>>(qkvh, route, attn, 1);

    // fused output projection: (65536 x 256) @ (128 x 256)^T -> f32 d_out
    gemm_wmma_kernel<1><<<dim3(NTOK / 128, 1), 256, 0, stream>>>(
        attn, Wch, bo, d_out, NTOK, 128, 256);
}
